// Model_76553497084046
// MI455X (gfx1250) — matrix operand order repaired and checked by reading, not compiled, not run on hardware
//
#include <hip/hip_runtime.h>
#include <hip/hip_bf16.h>
#include <stdint.h>

// ---------------- types ----------------
typedef __attribute__((ext_vector_type(16))) __bf16 v16bf;
typedef __attribute__((ext_vector_type(8)))  float  v8f;
typedef unsigned short u16;
typedef unsigned int   u32;
// Explicit global-address-space pointer: keeps the laundered weight pointer
// lowering to global_load_b128 (LOADcnt only) instead of flat_load_b128
// (which ties up DScnt and the LDS arbitration path too).
typedef const __attribute__((address_space(1))) u32* gwptr;

// ---------------- problem constants ----------------
constexpr int kV  = 400000;
constexpr int kE  = 100;
constexpr int kH  = 100;
constexpr int kB  = 4096;
constexpr int kT  = 32;
constexpr int kG3 = 300;        // 3*H
constexpr int kKP = 128;        // padded K (E and H both pad 100->128)
constexpr int kNT = 19;         // 19*16 = 304 >= 300 output-column tiles
constexpr int kKC = 4;          // 4 K-chunks of 32
constexpr int kBT = 16;         // batch rows per block
constexpr int kTHREADS = kNT * 32;          // 608 threads = 19 waves
constexpr int kFRAG_DW = kNT * kKC * 32 * 8; // dwords per packed weight matrix

// ---------------- helpers ----------------
__device__ __forceinline__ u16 f2bf(float f) {
  u32 u = __float_as_uint(f);
  u32 r = u + 0x7FFFu + ((u >> 16) & 1u);   // round-to-nearest-even
  return (u16)(r >> 16);
}

__device__ __forceinline__ float sigmoidf_fast(float x) {
  return 1.0f / (1.0f + __expf(-x));
}

// A-fragment (16x32 bf16, MxK) from LDS, per CDNA5 16-bit A layout:
// lanes 0-15: row M=lane, K = 0..7 (v0..3) and 16..23 (v4..7)
// lanes 16-31: row M=lane-16, K = 8..15 and 24..31
__device__ __forceinline__ v16bf load_a_frag(const u16* base, int stride,
                                             int kc, int lane) {
  const int hlf = lane >> 4;
  const int m   = lane & 15;
  const u16* row = base + m * stride + kc * 32;
  union { v16bf v; u16 u[16]; } r;
#pragma unroll
  for (int vv = 0; vv < 4; ++vv) {
    int kk = (hlf ? 8 : 0) + 2 * vv;
    r.u[2 * vv]     = row[kk];
    r.u[2 * vv + 1] = row[kk + 1];
  }
#pragma unroll
  for (int vv = 4; vv < 8; ++vv) {
    int kk = (hlf ? 24 : 16) + 2 * (vv - 4);
    r.u[2 * vv]     = row[kk];
    r.u[2 * vv + 1] = row[kk + 1];
  }
  return r.v;
}

// B-fragment: 8 contiguous dwords per lane from the pre-packed weight buffer.
__device__ __forceinline__ v16bf load_b_frag(gwptr p) {
  union { v16bf v; u32 u[8]; } r;
#pragma unroll
  for (int i = 0; i < 8; ++i) r.u[i] = p[i];
  return r.v;
}

// One 16x16 output tile of  out = A[16xKP] * Wpacked + bias
__device__ __forceinline__ v8f gemm_tile(const u16* aLds, int strideA,
                                         gwptr wmat, int nt, int lane,
                                         float bias) {
  v8f c;
#pragma unroll
  for (int i = 0; i < 8; ++i) c[i] = bias;
#pragma unroll
  for (int kc = 0; kc < kKC; ++kc) {
    v16bf a = load_a_frag(aLds, strideA, kc, lane);
    v16bf b = load_b_frag(wmat + ((size_t)(nt * kKC + kc) * 32 + lane) * 8);
    c = __builtin_amdgcn_wmma_f32_16x16x32_bf16(false, a, false, b,
                                                (short)0, c, false, false);
  }
  return c;
}

// ---------------- weight repack kernel ----------------
// Packs w[3H x 100] (row-major, fp32) into the wave32 WMMA B-matrix fragment
// layout (bf16): frag[nt][kc][lane][8 dwords]; lanes 0-15 hold K=0..7,16..23 of the
// chunk (2 per dword), lanes 16-31 hold K=8..15,24..31; column N = nt*16 + lane%16.
__device__ __forceinline__ u16 wfetch(const float* w, int n, int k) {
  return (n < kG3 && k < kE) ? f2bf(w[n * kE + k]) : (u16)0;
}

__global__ void pack_weights(const float* __restrict__ w0,
                             const float* __restrict__ w1,
                             const float* __restrict__ w2,
                             const float* __restrict__ w3,
                             u32* __restrict__ dst) {
  const int mat = blockIdx.y;
  const float* w = (mat == 0) ? w0 : (mat == 1) ? w1 : (mat == 2) ? w2 : w3;
  int tid = blockIdx.x * blockDim.x + threadIdx.x;
  if (tid >= kNT * kKC * 32) return;
  const int lane = tid & 31;
  const int kc   = (tid >> 5) & 3;
  const int nt   = tid >> 7;
  const int hlf  = lane >> 4;
  const int n    = nt * 16 + (lane & 15);
  u32* out = dst + (size_t)mat * kFRAG_DW + (size_t)tid * 8;
#pragma unroll
  for (int vv = 0; vv < 8; ++vv) {
    int k0 = kc * 32 + (vv < 4 ? 0 : 8) + hlf * 8 + 2 * vv;
    out[vv] = (u32)wfetch(w, n, k0) | ((u32)wfetch(w, n, k0 + 1) << 16);
  }
}

// ---------------- fused 2-layer GRU + maxpool + FC kernel ----------------
__global__ __launch_bounds__(kTHREADS)
void gru_fused(const int*   __restrict__ x,
               const float* __restrict__ emb,
               const float* __restrict__ h0in,
               const float* __restrict__ bih0, const float* __restrict__ bhh0,
               const float* __restrict__ bih1, const float* __restrict__ bhh1,
               const float* __restrict__ fcw,  const float* __restrict__ fcb,
               const u32*   __restrict__ wpack,
               float*       __restrict__ out) {
  // LDS state (all recurrence state lives here; no HBM round trips in t-loop)
  __shared__ u16   xbf [kBT][136];   // layer-0 input (bf16, padded K)
  __shared__ u16   h0bf[kBT][136];   // layer-0 hidden (bf16)
  __shared__ u16   h1bf[kBT][136];   // layer-1 hidden (bf16)
  __shared__ float h0f [kBT][104];   // layer-0 hidden (fp32)
  __shared__ float h1f [kBT][104];   // layer-1 hidden (fp32)
  __shared__ float S   [kBT][308];   // gi+gh, all 304 gate columns
  __shared__ float HN  [kBT][104];   // h_n alone (cols 200..303 -> 0..103)
  __shared__ float mx  [kBT][104];   // running max over time of layer-1 out
  __shared__ int   toks[kBT];

  const int tid   = threadIdx.x;
  const int lane  = tid & 31;
  const int wave  = tid >> 5;             // 0..18 -> output-column tile
  const int bbase = blockIdx.x * kBT;

  // ---- init hidden state from h0in (L,B,H), zero the K padding ----
  for (int i = tid; i < kBT * kKP; i += kTHREADS) {
    int m = i >> 7, j = i & 127;
    float v0 = (j < kH) ? h0in[(size_t)(bbase + m) * kH + j] : 0.0f;
    float v1 = (j < kH) ? h0in[(size_t)kB * kH + (size_t)(bbase + m) * kH + j]
                        : 0.0f;
    h0bf[m][j] = f2bf(v0);
    h1bf[m][j] = f2bf(v1);
  }
  for (int i = tid; i < kBT * kH; i += kTHREADS) {
    int m = i / kH, j = i % kH;
    h0f[m][j] = h0in[(size_t)(bbase + m) * kH + j];
    h1f[m][j] = h0in[(size_t)kB * kH + (size_t)(bbase + m) * kH + j];
    mx[m][j]  = -3.0e38f;
  }
  __syncthreads();

  const int hlf = lane >> 4;
  const int nl  = lane & 15;
  const int nn  = wave * 16 + nl;                 // global output column
  const bool ngate = (nn >= 200) && (nn < 304);   // columns feeding tanh gate
  const float bi0 = (nn < kG3) ? bih0[nn] : 0.0f;
  const float bh0 = (nn < kG3) ? bhh0[nn] : 0.0f;
  const float bi1 = (nn < kG3) ? bih1[nn] : 0.0f;
  const float bh1 = (nn < kG3) ? bhh1[nn] : 0.0f;

#pragma unroll 1
  for (int t = 0; t < kT; ++t) {
    // Launder the weight pointer through an opaque integer each iteration
    // (prevents hoisting all 16 B-fragments -> scratch spills), then cast
    // back into address_space(1) so the reloads stay global_load_b128.
    unsigned long long wpa = (unsigned long long)(uintptr_t)wpack;
    asm volatile("" : "+s"(wpa));
    gwptr wp = (gwptr)wpa;

    // ---- stage tokens; prefetch next step's embedding rows into caches ----
    if (tid < kBT) {
      toks[tid] = x[(size_t)(bbase + tid) * kT + t];
      if (t + 1 < kT) {
        int tn = x[(size_t)(bbase + tid) * kT + t + 1];
        __builtin_prefetch(&emb[(size_t)tn * kE], 0, 1); // global_prefetch_b8
      }
    }
    __syncthreads();

    // ---- gather embedding rows -> bf16 (vectorized: E=100 floats = 400 B,
    //      16-byte aligned rows, so float4 loads + uint2 LDS stores) ----
    if (tid < kBT * 32) {
      int m = tid >> 5;
      int c = (tid & 31) * 4;                  // 0,4,...,124
      float4 v = make_float4(0.0f, 0.0f, 0.0f, 0.0f);
      if (c < kE)                              // c=96 covers 96..99; c>=100 -> 0
        v = *(const float4*)(emb + (size_t)toks[m] * kE + c);
      union { uint2 d; u16 h[4]; } p;
      p.h[0] = f2bf(v.x); p.h[1] = f2bf(v.y);
      p.h[2] = f2bf(v.z); p.h[3] = f2bf(v.w);
      *(uint2*)&xbf[m][c] = p.d;               // ds_store_b64
    }
    __syncthreads();

    // ---- layer 0: gi = x@Wih0^T + bih0 ; gh = h0@Whh0^T + bhh0 ----
    {
      v8f cgi = gemm_tile(&xbf[0][0],  136, wp + 0 * (size_t)kFRAG_DW,
                          wave, lane, bi0);
      v8f cgh = gemm_tile(&h0bf[0][0], 136, wp + 1 * (size_t)kFRAG_DW,
                          wave, lane, bh0);
#pragma unroll
      for (int vv = 0; vv < 8; ++vv)            // branch-free full-width store
        S[vv + hlf * 8][nn] = cgi[vv] + cgh[vv];
      if (ngate) {                              // single divergence region
#pragma unroll
        for (int vv = 0; vv < 8; ++vv)
          HN[vv + hlf * 8][nn - 200] = cgh[vv];
      }
    }
    __syncthreads();

    // ---- layer 0 gates + state update:  n = tanh(S_n + (r-1)*h_n) ----
    for (int i = tid; i < kBT * kH; i += kTHREADS) {
      int m = i / kH, j = i % kH;
      float r  = sigmoidf_fast(S[m][j]);
      float z  = sigmoidf_fast(S[m][100 + j]);
      float nv = tanhf(S[m][200 + j] + (r - 1.0f) * HN[m][j]);
      float hn = (1.0f - z) * nv + z * h0f[m][j];
      h0f[m][j]  = hn;
      h0bf[m][j] = f2bf(hn);
    }
    __syncthreads();

    // ---- layer 1: gi = h0new@Wih1^T + bih1 ; gh = h1@Whh1^T + bhh1 ----
    {
      v8f cgi = gemm_tile(&h0bf[0][0], 136, wp + 2 * (size_t)kFRAG_DW,
                          wave, lane, bi1);
      v8f cgh = gemm_tile(&h1bf[0][0], 136, wp + 3 * (size_t)kFRAG_DW,
                          wave, lane, bh1);
#pragma unroll
      for (int vv = 0; vv < 8; ++vv)
        S[vv + hlf * 8][nn] = cgi[vv] + cgh[vv];
      if (ngate) {
#pragma unroll
        for (int vv = 0; vv < 8; ++vv)
          HN[vv + hlf * 8][nn - 200] = cgh[vv];
      }
    }
    __syncthreads();

    // ---- layer 1 gates + state update + running max (fused MaxPool) ----
    for (int i = tid; i < kBT * kH; i += kTHREADS) {
      int m = i / kH, j = i % kH;
      float r  = sigmoidf_fast(S[m][j]);
      float z  = sigmoidf_fast(S[m][100 + j]);
      float nv = tanhf(S[m][200 + j] + (r - 1.0f) * HN[m][j]);
      float hn = (1.0f - z) * nv + z * h1f[m][j];
      h1f[m][j]  = hn;
      h1bf[m][j] = f2bf(hn);
      mx[m][j]   = fmaxf(mx[m][j], hn);
    }
    __syncthreads();
  }

  // ---- final FC + sigmoid: one wave per batch row, shuffle reduction ----
  if (tid < kBT * 32) {
    int m = tid >> 5, l = tid & 31;
    float acc = 0.0f;
#pragma unroll
    for (int j = l; j < kH; j += 32) acc += fcw[j] * mx[m][j];
#pragma unroll
    for (int off = 16; off > 0; off >>= 1) acc += __shfl_down(acc, off, 32);
    if (l == 0) out[bbase + m] = sigmoidf_fast(acc + fcb[0]);
  }
}

// ---------------- launcher ----------------
extern "C" void kernel_launch(void* const* d_in, const int* in_sizes, int n_in,
                              void* d_out, int out_size, void* d_ws,
                              size_t ws_size, hipStream_t stream) {
  (void)in_sizes; (void)n_in; (void)out_size; (void)ws_size;
  const int*   x    = (const int*)  d_in[0];
  const float* emb  = (const float*)d_in[1];
  const float* h0   = (const float*)d_in[2];
  const float* wih0 = (const float*)d_in[3];
  const float* whh0 = (const float*)d_in[4];
  const float* bih0 = (const float*)d_in[5];
  const float* bhh0 = (const float*)d_in[6];
  const float* bih1 = (const float*)d_in[9];
  const float* whh1 = (const float*)d_in[8];
  const float* wih1 = (const float*)d_in[7];
  const float* bhh1 = (const float*)d_in[10];
  const float* fcw  = (const float*)d_in[11];
  const float* fcb  = (const float*)d_in[12];
  u32* wpack = (u32*)d_ws;   // 4 * 77824 B = 311296 B of packed bf16 weights

  dim3 pgrid((kNT * kKC * 32 + 255) / 256, 4);
  pack_weights<<<pgrid, 256, 0, stream>>>(wih0, whh0, wih1, whh1, wpack);

  gru_fused<<<kB / kBT, kTHREADS, 0, stream>>>(
      x, emb, h0, bih0, bhh0, bih1, bhh1, fcw, fcb, wpack, (float*)d_out);
}
